// SimpleRNN_53008486367293
// MI455X (gfx1250) — matrix operand order repaired and checked by reading, not compiled, not run on hardware
//
#include <hip/hip_runtime.h>
#include <math.h>

// SimpleRNN for MI455X (gfx1250, wave32, WMMA bf16 -> f32 accum)
//   x:    [128, 512, 256] f32      kernel W:  [256, 512] f32
//   Wh:   [512, 512] f32           bias:      [512] f32
//   out:  [128, 512, 512] f32  (doubles as xp scratch, overwritten in place)
//
// Design:
//  * Weights pre-packed once per launch into a contiguous-k (k = 16*half + j) WMMA B-operand
//    per-lane bf16 layout (device globals) -> hot-loop B = contiguous v16bf.
//  * 1024-thread blocks (32 waves = all 32 N-tiles of one M-tile). The A tile
//    is cooperatively converted to bf16 and staged in LDS in a contiguous-k (k = 16*half + s) WMMA
//    A-operand per-lane dword layout [kc][dword][lane] (lane == bank =>
//    conflict-free ds_load_b32). Each x / h_prev element is read from global
//    exactly once per block.

#define BATCH  128
#define SEQ    512
#define IN_DIM 256
#define UNITS  512

typedef __attribute__((ext_vector_type(16))) __bf16 v16bf;
typedef __attribute__((ext_vector_type(8)))  float  v8f;
typedef unsigned int u32;

__device__ __forceinline__ __bf16 f2bf(float f) { return (__bf16)f; }

// Pack two f32 into one dword of two bf16 (lo = first). Lowers to
// v_cvt_pk_bf16_f32.
__device__ __forceinline__ u32 pk2(float lo, float hi) {
    union { __bf16 h[2]; u32 u; } r;
    r.h[0] = (__bf16)lo; r.h[1] = (__bf16)hi;
    return r.u;
}

// Packed B layout: [nt][kchunk][lane][16 bf16], 32B per lane per chunk.
// Element j of lane L (half = L>>4, n = nt*16 + (L&15)) holds
//   B[kchunk*32 + 16*half + j][n]  (lanes 0-15: K 0-15, lanes 16-31: K 16-31)
__device__ __bf16 g_wh_packed[(UNITS / 16) * (UNITS / 32) * 32 * 16];   // 512 KB
__device__ __bf16 g_w_packed [(UNITS / 16) * (IN_DIM / 32) * 32 * 16];  // 256 KB

// ---------------------------------------------------------------------------
// One-time weight packing (runs every kernel_launch; deterministic).
// ---------------------------------------------------------------------------
__global__ void pack_wh_kernel(const float* __restrict__ Wh) {
    const int tid  = blockIdx.x * blockDim.x + threadIdx.x;
    const int lane = tid & 31;
    const int wave = tid >> 5;                 // nt*16 + kc, 512 waves
    const int kc   = wave & 15;
    const int nt   = wave >> 4;
    const int n    = nt * 16 + (lane & 15);
    const int kb   = kc * 32 + 16 * (lane >> 4);
    v16bf v;
#pragma unroll
    for (int j = 0; j < 16; ++j) v[j] = f2bf(Wh[(size_t)(kb + j) * UNITS + n]);
    *(v16bf*)&g_wh_packed[(size_t)(wave * 32 + lane) * 16] = v;
}

__global__ void pack_w_kernel(const float* __restrict__ W) {
    const int tid  = blockIdx.x * blockDim.x + threadIdx.x;
    const int lane = tid & 31;
    const int wave = tid >> 5;                 // nt*8 + kc, 256 waves
    const int kc   = wave & 7;
    const int nt   = wave >> 3;
    const int n    = nt * 16 + (lane & 15);
    const int kb   = kc * 32 + 16 * (lane >> 4);
    v16bf v;
#pragma unroll
    for (int j = 0; j < 16; ++j) v[j] = f2bf(W[(size_t)(kb + j) * UNITS + n]);
    *(v16bf*)&g_w_packed[(size_t)(wave * 32 + lane) * 16] = v;
}

// ---------------------------------------------------------------------------
// Phase 1: xp = x2d @ W + bias, written directly into d_out.
//   x2d: [65536, 256]; out flat m*UNITS+n == out[b][t][n] with m = b*SEQ+t.
// Block = 1024 threads = 32 waves; block owns 64 rows (4 M-subtiles), wave w
// owns N-tile nt = w. A (64x256 f32) staged once in LDS as bf16 (32 KB).
// ---------------------------------------------------------------------------
__global__ __launch_bounds__(1024)
void rnn_input_proj_kernel(const float* __restrict__ x,
                           const float* __restrict__ bias,
                           float* __restrict__ out) {
    __shared__ u32 lds_a[4 * 8 * 8 * 32];     // [i][kc][d][lane] dwords, 32 KB

    const int tid  = threadIdx.x;
    const int lane = tid & 31;
    const int nt   = tid >> 5;                // wave id == N-tile (0..31)
    const int mb   = blockIdx.x * 64;

    // ---- cooperative A stage: one (i, kc, lane) slot per thread (1024 slots)
    {
        const int l  = tid & 31;
        const int kc = (tid >> 5) & 7;
        const int i  = tid >> 8;              // 0..3
        const int row = mb + i * 16 + (l & 15);
        const int kb  = kc * 32 + 16 * (l >> 4);
        const float* p = x + (size_t)row * IN_DIM + kb;
        u32* dst = &lds_a[((i * 8 + kc) * 8) * 32 + l];
#pragma unroll
        for (int d = 0; d < 4; ++d) dst[d * 32]       = pk2(p[2*d],      p[2*d + 1]);
#pragma unroll
        for (int d = 0; d < 4; ++d) dst[(4 + d) * 32] = pk2(p[8 + 2*d], p[9 + 2*d]);
    }
    __syncthreads();

    v8f acc0 = {}, acc1 = {}, acc2 = {}, acc3 = {};
    for (int kc = 0; kc < IN_DIM / 32; ++kc) {
        const v16bf b = *(const v16bf*)
            &g_w_packed[(size_t)((nt * (IN_DIM / 32) + kc) * 32 + lane) * 16];
#pragma unroll
        for (int i = 0; i < 4; ++i) {
            union { v16bf v; u32 u[8]; } a;
            const u32* src = &lds_a[((i * 8 + kc) * 8) * 32 + lane];
#pragma unroll
            for (int d = 0; d < 8; ++d) a.u[d] = src[d * 32];
            v8f* accp = (i == 0) ? &acc0 : (i == 1) ? &acc1 : (i == 2) ? &acc2 : &acc3;
            *accp = __builtin_amdgcn_wmma_f32_16x16x32_bf16(
                false, a.v, false, b, (short)0, *accp, false, false);
        }
    }

    const int half = lane >> 4;
    const int n    = nt * 16 + (lane & 15);
    const float bv = bias[n];
#pragma unroll
    for (int i = 0; i < 4; ++i) {
        const v8f acc = (i == 0) ? acc0 : (i == 1) ? acc1 : (i == 2) ? acc2 : acc3;
#pragma unroll
        for (int r = 0; r < 8; ++r) {
            const int mm = mb + i * 16 + half * 8 + r;   // C/D: M = 8*half + r
            out[(size_t)mm * UNITS + n] = acc[r] + bv;
        }
    }
}

// ---------------------------------------------------------------------------
// Phase 2 (one launch per timestep t): h_t = tanh(xp_t + h_{t-1} @ Wh).
// Block = 1024 threads = 32 waves = all 32 N-tiles of one batch M-tile; 8
// blocks per step. h_prev tile (16x512 f32) staged once in LDS as bf16
// (16 KB); packed Wh (512 KB bf16) stays L2-resident across all 512 steps.
// ---------------------------------------------------------------------------
__global__ __launch_bounds__(1024)
void rnn_step_kernel(float* __restrict__ out, int t) {
    __shared__ u32 lds_a[16 * 8 * 32];        // [kc][d][lane] dwords, 16 KB

    const int tid  = threadIdx.x;
    const int lane = tid & 31;
    const int nt   = tid >> 5;                // wave id == N-tile (0..31)
    const int mt   = blockIdx.x;              // batch tile (0..7)

    v8f acc = {};
    if (t > 0) {
        // ---- cooperative h_prev stage: 512 (kc, lane) slots
        if (tid < 512) {
            const int l  = tid & 31;
            const int kc = tid >> 5;          // 0..15
            const int row = mt * 16 + (l & 15);
            const int kb  = kc * 32 + 16 * (l >> 4);
            const float* p = out + (size_t)row * SEQ * UNITS
                                 + (size_t)(t - 1) * UNITS + kb;
            u32* dst = &lds_a[(kc * 8) * 32 + l];
#pragma unroll
            for (int d = 0; d < 4; ++d) dst[d * 32]       = pk2(p[2*d],      p[2*d + 1]);
#pragma unroll
            for (int d = 0; d < 4; ++d) dst[(4 + d) * 32] = pk2(p[8 + 2*d], p[9 + 2*d]);
        }
        __syncthreads();

        for (int kc = 0; kc < UNITS / 32; ++kc) {
            const v16bf b = *(const v16bf*)
                &g_wh_packed[(size_t)((nt * (UNITS / 32) + kc) * 32 + lane) * 16];
            union { v16bf v; u32 u[8]; } a;
            const u32* src = &lds_a[(kc * 8) * 32 + lane];
#pragma unroll
            for (int d = 0; d < 8; ++d) a.u[d] = src[d * 32];
            acc = __builtin_amdgcn_wmma_f32_16x16x32_bf16(
                false, a.v, false, b, (short)0, acc, false, false);
        }
    }

    const int half = lane >> 4;
    const int n    = nt * 16 + (lane & 15);
    const size_t t_off = (size_t)t * UNITS;
#pragma unroll
    for (int r = 0; r < 8; ++r) {
        const int bb = mt * 16 + half * 8 + r;           // batch index
        const size_t idx = (size_t)bb * SEQ * UNITS + t_off + n;
        out[idx] = tanhf(out[idx] + acc[r]);             // read xp, write h_t
    }
}

extern "C" void kernel_launch(void* const* d_in, const int* in_sizes, int n_in,
                              void* d_out, int out_size, void* d_ws, size_t ws_size,
                              hipStream_t stream) {
    (void)in_sizes; (void)n_in; (void)out_size; (void)d_ws; (void)ws_size;
    const float* x    = (const float*)d_in[0];   // [128, 512, 256]
    const float* W    = (const float*)d_in[1];   // [256, 512]
    const float* Wh   = (const float*)d_in[2];   // [512, 512]
    const float* bias = (const float*)d_in[3];   // [512]
    float* out = (float*)d_out;                  // [128, 512, 512]

    // Pack weights into contiguous-k bf16 WMMA B-operand layout (device globals).
    pack_wh_kernel<<<64, 256, 0, stream>>>(Wh);  // 512 waves
    pack_w_kernel <<<32, 256, 0, stream>>>(W);   // 256 waves

    // Phase 1: 1024 blocks x 32 waves; each block computes 64 x 512 outputs.
    rnn_input_proj_kernel<<<BATCH * SEQ / 64, 1024, 0, stream>>>(x, bias, out);

    // Phase 2: 512 dependent steps; 8 blocks x 32 waves per step.
    for (int t = 0; t < SEQ; ++t) {
        rnn_step_kernel<<<BATCH / 16, 1024, 0, stream>>>(out, t);
    }
}
